// RoPEMultiheadAttention_24438363914817
// MI455X (gfx1250) — matrix operand order repaired and checked by reading, not compiled, not run on hardware
//
#include <hip/hip_runtime.h>
#include <stdint.h>

// ---------------------------------------------------------------------------
// RoPE multihead attention for MI455X (gfx1250), bf16 WMMA datapath with
// async global->LDS copies (ASYNCcnt) and double-buffered LDS tiles.
// B=2, S=2048, D=1024, H=16, hd=64.
// ---------------------------------------------------------------------------

typedef __attribute__((ext_vector_type(16))) __bf16 bf16x16;
typedef __attribute__((ext_vector_type(8)))  float  f32x8;

#define WMMA_BF16(A_, B_, C_) \
  __builtin_amdgcn_wmma_f32_16x16x32_bf16(false, (A_), false, (B_), (short)0, (C_), false, false)

// Async DMA: global memory -> LDS, 16B per lane, tracked by ASYNCcnt.
// LDS operand is the byte offset = low 32 bits of the generic pointer
// (LDS aperture: ADDR[31:0] is the LDS address).
__device__ __forceinline__ void async_ld_b128(void* lds, const void* gaddr) {
  asm volatile("global_load_async_to_lds_b128 %0, %1, off"
               :: "v"((uint32_t)(uintptr_t)lds), "v"(gaddr)
               : "memory");
}
__device__ __forceinline__ void wait_async0() {
  asm volatile("s_wait_asynccnt 0x0" ::: "memory");
}

__device__ __forceinline__ __bf16 f2bf(float f) {
  union { float f; uint32_t u; } c; c.f = f;
  uint32_t r = c.u + 0x7FFFu + ((c.u >> 16) & 1u);   // round-to-nearest-even
  uint16_t h = (uint16_t)(r >> 16);
  return __builtin_bit_cast(__bf16, h);
}

// A-fragment (16x32 bf16, M x K): lane&15 = row M, lane>>4 = half.
// element e -> K = e + 16*half   (16 contiguous K values)
__device__ __forceinline__ bf16x16 ldfragA(const __bf16* base, int stride, int lane) {
  const __bf16* p = base + (lane & 15) * stride + ((lane >> 4) << 4);
  bf16x16 f;
#pragma unroll
  for (int q = 0; q < 8; ++q) {
    int k = q << 1;
    f[2 * q]     = p[k];
    f[2 * q + 1] = p[k + 1];
  }
  return f;
}

// B-fragment (32x16 bf16, K x N) from an [N][K] row-major array:
// lane&15 = column N, element e -> K = e + 16*half  (16 contiguous K values)
__device__ __forceinline__ bf16x16 ldfragB(const __bf16* base, int stride, int lane) {
  const __bf16* p = base + (lane & 15) * stride + ((lane >> 4) << 4);
  bf16x16 f;
#pragma unroll
  for (int e = 0; e < 16; ++e) f[e] = p[e];
  return f;
}

// B-fragment (32x16 bf16, K x N) from a [K][N] row-major array (strided):
__device__ __forceinline__ bf16x16 ldfragBT(const __bf16* base, int stride, int lane) {
  const __bf16* p = base + (lane & 15) + ((lane >> 4) << 4) * stride;
  bf16x16 f;
#pragma unroll
  for (int e = 0; e < 16; ++e) f[e] = p[e * stride];
  return f;
}

// ---------------------------------------------------------------------------
// GEMM: C[128x128] = A[128xK] * W[128 rows x K]^T, K = 1024, BK = 32.
// 8 waves: 2(M) x 4(N); each wave 64x32 = 4x2 C-fragments (8 WMMA / K-step).
// Double-buffered LDS tiles filled by async copies; one barrier / K-step.
// LDS rows padded to 40 bf16 (80B) -> 16B-aligned b128 transfers.
// ---------------------------------------------------------------------------
#define GBUF (128 * 40)

__device__ __forceinline__ void gemm_tile_copy(
    const __bf16* __restrict__ A, const __bf16* __restrict__ W,
    __bf16* As, __bf16* Bs, int m0, int n0, int kt, int t) {
  int row = t >> 1;              // 0..127
  int seg = (t & 1) << 4;        // 0 or 16 bf16 elems (2 x b128 each)
  const __bf16* ga = A + (size_t)(m0 + row) * 1024 + kt + seg;
  const __bf16* gb = W + (size_t)(n0 + row) * 1024 + kt + seg;
  __bf16* la = As + row * 40 + seg;
  __bf16* lb = Bs + row * 40 + seg;
  async_ld_b128(la,     ga);
  async_ld_b128(la + 8, ga + 8);
  async_ld_b128(lb,     gb);
  async_ld_b128(lb + 8, gb + 8);
}

__device__ __forceinline__ void gemm_mainloop(
    const __bf16* __restrict__ A, const __bf16* __restrict__ W,
    __bf16* As, __bf16* Bs, int m0, int n0, f32x8 acc[4][2]) {
  const int t = threadIdx.x, lane = t & 31, w = t >> 5;
  const int wm = (w >> 2) * 64, wn = (w & 3) * 32;
  gemm_tile_copy(A, W, As, Bs, m0, n0, 0, t);            // prologue: tile 0
  for (int kt = 0; kt < 1024; kt += 32) {
    const int cur = (kt >> 5) & 1;
    wait_async0();            // own copies for tile kt done
    __syncthreads();          // everyone's copies done; prev compute done
    if (kt + 32 < 1024)
      gemm_tile_copy(A, W, As + (cur ^ 1) * GBUF, Bs + (cur ^ 1) * GBUF,
                     m0, n0, kt + 32, t);                // overlap next tile
    const __bf16* at = As + cur * GBUF;
    const __bf16* bt = Bs + cur * GBUF;
    bf16x16 af[4], bw[2];
#pragma unroll
    for (int i = 0; i < 4; ++i) af[i] = ldfragA(at + (wm + i * 16) * 40, 40, lane);
#pragma unroll
    for (int j = 0; j < 2; ++j) bw[j] = ldfragB(bt + (wn + j * 16) * 40, 40, lane);
#pragma unroll
    for (int i = 0; i < 4; ++i)
#pragma unroll
      for (int j = 0; j < 2; ++j)
        acc[i][j] = WMMA_BF16(af[i], bw[j], acc[i][j]);
  }
  __syncthreads();
}

// ---------------------------------------------------------------------------
// f32 -> bf16 conversion
// ---------------------------------------------------------------------------
__global__ __launch_bounds__(256) void cvt_f32_to_bf16(
    const float* __restrict__ src, __bf16* __restrict__ dst, int n) {
  int i = blockIdx.x * blockDim.x + threadIdx.x;
  int stride = gridDim.x * blockDim.x;
  for (; i < n; i += stride) dst[i] = f2bf(src[i]);
}

// ---------------------------------------------------------------------------
// Fused QKV projection:  Y[4096 x 4096] = Xb * Wcat^T + bias,
// epilogue: RoPE(q,k) via lane shuffle, scatter to (B,H,S,hd) layouts.
// Wcat rows: [0,2048)=Wq, [2048,3072)=Wk, [3072,4096)=Wv.
// ---------------------------------------------------------------------------
__global__ __launch_bounds__(256) void qkv_gemm_rope_kernel(
    const __bf16* __restrict__ Xb, const __bf16* __restrict__ Wcat,
    const float* __restrict__ bq, const float* __restrict__ bkp,
    const float* __restrict__ bvp,
    __bf16* __restrict__ qb, __bf16* __restrict__ kb, __bf16* __restrict__ vbuf,
    float* __restrict__ gate) {
  __shared__ __align__(16) __bf16 As[2 * GBUF];
  __shared__ __align__(16) __bf16 Bs[2 * GBUF];
  const int m0 = blockIdx.y * 128, n0 = blockIdx.x * 128;
  f32x8 acc[4][2] = {};
  gemm_mainloop(Xb, Wcat, As, Bs, m0, n0, acc);

  const int t = threadIdx.x, lane = t & 31, w = t >> 5;
  const int wm = (w >> 2) * 64, wn = (w & 3) * 32;
  const float LN1E4_64 = 0.14391156463f;  // ln(10000)/64
#pragma unroll
  for (int i = 0; i < 4; ++i) {
#pragma unroll
    for (int j = 0; j < 2; ++j) {
#pragma unroll
      for (int r = 0; r < 8; ++r) {
        int m = m0 + wm + i * 16 + r + ((lane >> 4) << 3);
        int n = n0 + wn + j * 16 + (lane & 15);
        float bias = (n < 2048) ? bq[n] : (n < 3072 ? bkp[n - 2048] : bvp[n - 3072]);
        float v = acc[i][j][r] + bias;
        // RoPE partner (dim pair 2i/2i+1) lives in the adjacent lane.
        float other = __shfl_xor(v, 1, 32);
        int b = m >> 11, s = m & 2047;
        if (n < 2048) {                       // q_raw region
          int hh = n >> 7, c = n & 127;
          if (c < 64) {                       // q half -> RoPE
            float fr = __expf(-(float)(c & ~1) * LN1E4_64);
            float sn, cs; __sincosf((float)s * fr, &sn, &cs);
            float o = (c & 1) ? (v * cs + other * sn) : (v * cs - other * sn);
            qb[(((size_t)(b * 16 + hh) * 2048 + s) << 6) + c] = f2bf(o);
          } else {                            // gate half (f32, used later)
            gate[(((size_t)(b * 16 + hh) * 2048 + s) << 6) + (c - 64)] = v;
          }
        } else if (n < 3072) {                // k -> RoPE
          int nk = n - 2048, hh = nk >> 6, c = nk & 63;
          float fr = __expf(-(float)(c & ~1) * LN1E4_64);
          float sn, cs; __sincosf((float)s * fr, &sn, &cs);
          float o = (c & 1) ? (v * cs + other * sn) : (v * cs - other * sn);
          kb[(((size_t)(b * 16 + hh) * 2048 + s) << 6) + c] = f2bf(o);
        } else {                              // v
          int nv = n - 3072, hh = nv >> 6, c = nv & 63;
          vbuf[(((size_t)(b * 16 + hh) * 2048 + s) << 6) + c] = f2bf(v);
        }
      }
    }
  }
}

// ---------------------------------------------------------------------------
// Flash attention: block = (b, h, 128 query rows); 8 waves x 16 rows.
// Key chunks of 64, K and V async-copied row-major into double-buffered LDS;
// online softmax; P restaged via per-wave LDS (C-layout -> A-layout);
// V B-fragments gathered strided from the row-major [key][dim] tile.
// ---------------------------------------------------------------------------
#define ABUF (64 * 72)

__device__ __forceinline__ void attn_tile_copy(
    const __bf16* __restrict__ kg, const __bf16* __restrict__ vg,
    __bf16* ksb, __bf16* vsb, int t) {
  int row = t >> 2;              // 0..63
  int seg = (t & 3) << 4;        // 0,16,32,48 bf16 elems (2 x b128 each)
  const __bf16* gk = kg + row * 64 + seg;
  const __bf16* gv = vg + row * 64 + seg;
  __bf16* lk = ksb + row * 72 + seg;
  __bf16* lv = vsb + row * 72 + seg;
  async_ld_b128(lk,     gk);
  async_ld_b128(lk + 8, gk + 8);
  async_ld_b128(lv,     gv);
  async_ld_b128(lv + 8, gv + 8);
}

__global__ __launch_bounds__(256) void attn_kernel(
    const __bf16* __restrict__ qb, const __bf16* __restrict__ kb,
    const __bf16* __restrict__ vbuf, const float* __restrict__ gate,
    __bf16* __restrict__ ag) {
  __shared__ __align__(16) __bf16 ks[2 * ABUF];       // [key][dim] x2
  __shared__ __align__(16) __bf16 vs[2 * ABUF];       // [key][dim] x2
  __shared__ __align__(16) __bf16 ps[8 * 16 * 72];    // per-wave P staging
  const int t = threadIdx.x, lane = t & 31, w = t >> 5;
  const int qt = blockIdx.x, hh = blockIdx.y, b = blockIdx.z;
  const size_t bh = (size_t)(b * 16 + hh);
  const __bf16* kgb = kb   + bh * 2048 * 64;
  const __bf16* vgb = vbuf + bh * 2048 * 64;

  const __bf16* qbase = qb + (bh * 2048 + (size_t)(qt * 128 + w * 16)) * 64;
  bf16x16 aq0 = ldfragA(qbase, 64, lane);        // q dims 0..31
  bf16x16 aq1 = ldfragA(qbase + 32, 64, lane);   // q dims 32..63

  float mrun[8], lrun[8];
  f32x8 acco[4] = {};
#pragma unroll
  for (int r = 0; r < 8; ++r) { mrun[r] = -1e30f; lrun[r] = 0.0f; }
  __bf16* pw = &ps[w * 16 * 72];

  attn_tile_copy(kgb, vgb, ks, vs, t);                 // prologue: chunk 0
  for (int ch = 0; ch < 32; ++ch) {
    const int cur = ch & 1;
    wait_async0();
    __syncthreads();
    if (ch + 1 < 32)
      attn_tile_copy(kgb + (size_t)(ch + 1) * 64 * 64,
                     vgb + (size_t)(ch + 1) * 64 * 64,
                     ks + (cur ^ 1) * ABUF, vs + (cur ^ 1) * ABUF, t);
    const __bf16* kt_ = ks + cur * ABUF;
    const __bf16* vt_ = vs + cur * ABUF;

    // scores: 4 x (16q x 16k) tiles, hd=64 -> 2 WMMA each
    f32x8 sc[4];
#pragma unroll
    for (int j2 = 0; j2 < 4; ++j2) {
      bf16x16 bk0 = ldfragB(kt_ + (j2 * 16) * 72, 72, lane);
      bf16x16 bk1 = ldfragB(kt_ + (j2 * 16) * 72 + 32, 72, lane);
      f32x8 z = {};
      z = WMMA_BF16(aq0, bk0, z);
      z = WMMA_BF16(aq1, bk1, z);
      sc[j2] = z;
    }

    // online softmax (row r of fragment = q-row r + 8*(lane>=16))
#pragma unroll
    for (int r = 0; r < 8; ++r) {
      float mx = -1e30f;
#pragma unroll
      for (int j2 = 0; j2 < 4; ++j2) {
        sc[j2][r] *= 0.125f;                 // SCALE = hd^-0.5
        mx = fmaxf(mx, sc[j2][r]);
      }
#pragma unroll
      for (int off = 1; off < 16; off <<= 1) mx = fmaxf(mx, __shfl_xor(mx, off, 32));
      float mnew = fmaxf(mrun[r], mx);
      float fac = __expf(mrun[r] - mnew);
      mrun[r] = mnew;
      lrun[r] *= fac;
      float psum = 0.0f;
      int rowo = (r + ((lane >> 4) << 3)) * 72 + (lane & 15);
#pragma unroll
      for (int j2 = 0; j2 < 4; ++j2) {
        float p = __expf(sc[j2][r] - mnew);
        pw[rowo + j2 * 16] = f2bf(p);        // C-layout -> LDS
        psum += p;
      }
#pragma unroll
      for (int off = 1; off < 16; off <<= 1) psum += __shfl_xor(psum, off, 32);
      lrun[r] += psum;
#pragma unroll
      for (int nf = 0; nf < 4; ++nf) acco[nf][r] *= fac;
    }

    // P @ V  (keys 0..63 -> 2 A-frags; hd 64 -> 4 N-tiles)
    bf16x16 ap0 = ldfragA(pw, 72, lane);
    bf16x16 ap1 = ldfragA(pw + 32, 72, lane);
#pragma unroll
    for (int nf = 0; nf < 4; ++nf) {
      bf16x16 bv0 = ldfragBT(vt_ + nf * 16,           72, lane);  // keys 0..31
      bf16x16 bv1 = ldfragBT(vt_ + nf * 16 + 32 * 72, 72, lane);  // keys 32..63
      acco[nf] = WMMA_BF16(ap0, bv0, acco[nf]);
      acco[nf] = WMMA_BF16(ap1, bv1, acco[nf]);
    }
  }

  // normalize, sigmoid-gate, pack to [B,S,H*hd] bf16 for the output GEMM
#pragma unroll
  for (int nf = 0; nf < 4; ++nf) {
#pragma unroll
    for (int r = 0; r < 8; ++r) {
      int sq = qt * 128 + w * 16 + r + ((lane >> 4) << 3);
      int n = nf * 16 + (lane & 15);
      float o = acco[nf][r] / lrun[r];
      float g = gate[(bh * 2048 + sq) * 64 + n];
      o *= 1.0f / (1.0f + __expf(-g));
      ag[((size_t)(b * 2048 + sq) << 10) + hh * 64 + n] = f2bf(o);
    }
  }
}

// ---------------------------------------------------------------------------
// Output projection: out[4096 x 1024] = Ag * Wo^T + bo   (f32 out)
// ---------------------------------------------------------------------------
__global__ __launch_bounds__(256) void out_gemm_kernel(
    const __bf16* __restrict__ Ag, const __bf16* __restrict__ Wob,
    const float* __restrict__ bo, float* __restrict__ out) {
  __shared__ __align__(16) __bf16 As[2 * GBUF];
  __shared__ __align__(16) __bf16 Bs[2 * GBUF];
  const int m0 = blockIdx.y * 128, n0 = blockIdx.x * 128;
  f32x8 acc[4][2] = {};
  gemm_mainloop(Ag, Wob, As, Bs, m0, n0, acc);
  const int t = threadIdx.x, lane = t & 31, w = t >> 5;
  const int wm = (w >> 2) * 64, wn = (w & 3) * 32;
#pragma unroll
  for (int i = 0; i < 4; ++i)
#pragma unroll
    for (int j = 0; j < 2; ++j)
#pragma unroll
      for (int r = 0; r < 8; ++r) {
        int m = m0 + wm + i * 16 + r + ((lane >> 4) << 3);
        int n = n0 + wn + j * 16 + (lane & 15);
        out[(size_t)m * 1024 + n] = acc[i][j][r] + bo[n];
      }
}

// ---------------------------------------------------------------------------
// Host launcher. Workspace map (bytes):
//   [ 0, 8M)  Xb     query bf16            [ 8M,16M) Wcat  [Wq;Wk;Wv] bf16
//   [16M,18M) Wob    Wo bf16               [18M,26M) qb    (B,H,S,hd) bf16
//   [26M,34M) kb                            [34M,42M) vbuf
//   [42M,58M) gate   (B,H,S,hd) f32        [58M,66M) ag    (B,S,D) bf16
// ---------------------------------------------------------------------------
extern "C" void kernel_launch(void* const* d_in, const int* in_sizes, int n_in,
                              void* d_out, int out_size, void* d_ws, size_t ws_size,
                              hipStream_t stream) {
  const float* query = (const float*)d_in[0];
  const float* Wq = (const float*)d_in[1];
  const float* bq = (const float*)d_in[2];
  const float* Wk = (const float*)d_in[3];
  const float* bk = (const float*)d_in[4];
  const float* Wv = (const float*)d_in[5];
  const float* bv = (const float*)d_in[6];
  const float* Wo = (const float*)d_in[7];
  const float* bo = (const float*)d_in[8];
  float* out = (float*)d_out;

  char* ws = (char*)d_ws;
  __bf16* Xb   = (__bf16*)(ws);
  __bf16* Wcat = (__bf16*)(ws + (8ull  << 20));
  __bf16* Wob  = (__bf16*)(ws + (16ull << 20));
  __bf16* qb   = (__bf16*)(ws + (18ull << 20));
  __bf16* kb_  = (__bf16*)(ws + (26ull << 20));
  __bf16* vb_  = (__bf16*)(ws + (34ull << 20));
  float*  gate = (float*) (ws + (42ull << 20));
  __bf16* ag   = (__bf16*)(ws + (58ull << 20));

  dim3 blk(256);
  cvt_f32_to_bf16<<<2048, blk, 0, stream>>>(query, Xb, 4 * 1024 * 1024);
  cvt_f32_to_bf16<<<2048, blk, 0, stream>>>(Wq, Wcat, 2 * 1024 * 1024);
  cvt_f32_to_bf16<<<1024, blk, 0, stream>>>(Wk, Wcat + 2 * 1024 * 1024, 1024 * 1024);
  cvt_f32_to_bf16<<<1024, blk, 0, stream>>>(Wv, Wcat + 3 * 1024 * 1024, 1024 * 1024);
  cvt_f32_to_bf16<<<1024, blk, 0, stream>>>(Wo, Wob, 1024 * 1024);

  qkv_gemm_rope_kernel<<<dim3(32, 32), blk, 0, stream>>>(
      Xb, Wcat, bq, bk, bv, qb, kb_, vb_, gate);
  attn_kernel<<<dim3(16, 16, 2), blk, 0, stream>>>(qb, kb_, vb_, gate, ag);
  out_gemm_kernel<<<dim3(8, 32), blk, 0, stream>>>(ag, Wob, bo, out);

  (void)in_sizes; (void)n_in; (void)out_size; (void)ws_size;
}
